// GraphMamba_18425409699841
// MI455X (gfx1250) — matrix operand order repaired and checked by reading, not compiled, not run on hardware
//
#include <hip/hip_runtime.h>
#include <hip/hip_bf16.h>

// ---------------------------------------------------------------------------
// GraphMamba forward for MI455X (gfx1250, wave32, WMMA).
//  - GEMMs: bf16 WMMA (v_wmma_f32_16x16x32_bf16), fp32 accumulate, LDS-staged,
//    vectorized interior fast path (float4 loads, uint2 LDS stores, b128 gathers)
//  - GCN scatter: hardware fp32 global atomics
//  - Mamba selective scan: 3-pass chunked associative linear-recurrence scan
// ---------------------------------------------------------------------------

typedef __attribute__((ext_vector_type(16))) __bf16 v16bf;
typedef __attribute__((ext_vector_type(8)))  float  v8f;

namespace {
constexpr int Nn     = 10000;   // nodes / sequence length
constexpr int Ec     = 320000;  // edges
constexpr int Fc     = 128;     // input features
constexpr int Hc     = 256;     // hidden
constexpr int DIc    = 512;     // mamba d_inner
constexpr int DSc    = 16;      // mamba d_state
constexpr int NSc    = 10000;   // stations
constexpr int NCc    = 8;       // line colors
constexpr int NCHUNK = 100;     // scan chunks
constexpr int CLEN   = 100;     // steps per chunk (NCHUNK*CLEN == Nn)
}

// fp32 -> bf16 bits with round-to-nearest-even
__device__ __forceinline__ unsigned short bfbits(float f) {
    union { float f; unsigned u; } x; x.f = f;
    unsigned r = x.u + 0x7FFFu + ((x.u >> 16) & 1u);
    return (unsigned short)(r >> 16);
}
__device__ __forceinline__ unsigned pack2bf(float a, float b) {
    return (unsigned)bfbits(a) | ((unsigned)bfbits(b) << 16);
}

__device__ __forceinline__ void atomAddF(float* p, float v) {
    unsafeAtomicAdd(p, v);   // GLOBAL_ATOMIC_ADD_F32
}

// ---------------------------------------------------------------------------
// Generic bf16-WMMA GEMM:  C[M,N] = act(A[M,Kd] * B[Kd,N] + bias)
// Block = 128 threads (4 waves). Block tile 64x32 (2 WMMAs/wave/K-step),
// K-step 32. act: 0 = none, 1 = relu, 2 = softplus.
// Requires lda, ldb multiples of 4 (true for all call sites: 48..1024).
// ---------------------------------------------------------------------------
#define GT_M 64
#define GT_N 32
#define GT_K 32

__global__ __launch_bounds__(128) void k_gemm(
    const float* __restrict__ A, int lda,
    const float* __restrict__ B, int ldb,
    const float* __restrict__ bias,
    float* __restrict__ C, int ldc,
    int M, int Kd, int N, int act)
{
    __shared__ unsigned short As[GT_M][GT_K];   // bf16 bits, row-major (M x K)
    __shared__ unsigned short Bs[GT_N][GT_K];   // bf16 bits, transposed (N x K)

    const int tid  = threadIdx.x;
    const int lane = tid & 31;
    const int wave = tid >> 5;
    const int m0   = blockIdx.x * GT_M;
    const int n0   = blockIdx.y * GT_N;

    v8f acc0 = {}, acc1 = {};
    const bool mfull = (m0 + GT_M) <= M;
    const bool nfull = (n0 + GT_N) <= N;

    for (int k0 = 0; k0 < Kd; k0 += GT_K) {
        const bool interior = mfull && nfull && ((k0 + GT_K) <= Kd);
        if (interior) {
            // ---- fast path: float4 loads, packed bf16x2 LDS stores ----
            #pragma unroll
            for (int i = 0; i < 4; ++i) {           // A: 512 groups of 4
                int idx = i * 128 + tid;
                int r = idx >> 3, g = idx & 7;
                const float4 v = *reinterpret_cast<const float4*>(
                    A + (size_t)(m0 + r) * lda + k0 + g * 4);
                uint2 p; p.x = pack2bf(v.x, v.y); p.y = pack2bf(v.z, v.w);
                *reinterpret_cast<uint2*>(&As[r][g * 4]) = p;
            }
            #pragma unroll
            for (int i = 0; i < 2; ++i) {           // B: 256 groups of 4 cols
                int idx = i * 128 + tid;
                int kb = idx >> 3, g = idx & 7;
                const float4 v = *reinterpret_cast<const float4*>(
                    B + (size_t)(k0 + kb) * ldb + n0 + g * 4);
                Bs[g * 4 + 0][kb] = bfbits(v.x);
                Bs[g * 4 + 1][kb] = bfbits(v.y);
                Bs[g * 4 + 2][kb] = bfbits(v.z);
                Bs[g * 4 + 3][kb] = bfbits(v.w);
            }
            if (k0 + GT_K < Kd) {                   // prefetch next A K-tile
                __builtin_prefetch(
                    A + (size_t)(m0 + (tid & 63)) * lda + k0 + GT_K, 0, 1);
            }
        } else {
            // ---- guarded scalar path (edge blocks, Kd=16 dt_proj) ----
            #pragma unroll
            for (int i = 0; i < 16; ++i) {
                int idx = i * 128 + tid;
                int r = idx >> 5, kk = idx & 31;
                int row = m0 + r, k = k0 + kk;
                float v = (row < M && k < Kd) ? A[(size_t)row * lda + k] : 0.0f;
                As[r][kk] = bfbits(v);
            }
            #pragma unroll
            for (int i = 0; i < 8; ++i) {
                int idx = i * 128 + tid;
                int nn = idx >> 5, kb = idx & 31;
                int k = k0 + kb, col = n0 + nn;
                float v = (k < Kd && col < N) ? B[(size_t)k * ldb + col] : 0.0f;
                Bs[nn][kb] = bfbits(v);
            }
        }
        __syncthreads();

        // ---- A fragment (CDNA5 16-bit striping): lane<16 -> K {0..7,16..23},
        //      lane>=16 -> K {8..15,24..31}; two 16B runs -> ds_load_b128 x2
        const int arow  = wave * 16 + (lane & 15);
        const int abase = (lane < 16) ? 0 : 8;
        union { uint4 q[2]; v16bf v; } ua;
        ua.q[0] = *reinterpret_cast<const uint4*>(&As[arow][abase]);
        ua.q[1] = *reinterpret_cast<const uint4*>(&As[arow][abase + 16]);

        // ---- B fragments: lane<16 -> K {0..7,16..23}, lane>=16 -> K {8..15,24..31}
        const int bn    = lane & 15;
        const int bbase = (lane < 16) ? 0 : 8;
        union { uint4 q[2]; v16bf v; } ub0, ub1;
        ub0.q[0] = *reinterpret_cast<const uint4*>(&Bs[bn][bbase]);
        ub0.q[1] = *reinterpret_cast<const uint4*>(&Bs[bn][bbase + 16]);
        ub1.q[0] = *reinterpret_cast<const uint4*>(&Bs[bn + 16][bbase]);
        ub1.q[1] = *reinterpret_cast<const uint4*>(&Bs[bn + 16][bbase + 16]);

        acc0 = __builtin_amdgcn_wmma_f32_16x16x32_bf16(
            false, ua.v, false, ub0.v, (short)0, acc0, false, false);
        acc1 = __builtin_amdgcn_wmma_f32_16x16x32_bf16(
            false, ua.v, false, ub1.v, (short)0, acc1, false, false);
        __syncthreads();
    }

    // ---- writeback: VGPR r -> M=r (lanes<16) / M=8+r (lanes>=16), N=lane&15
    const int rhi = (lane >> 4) & 1;
    #pragma unroll
    for (int r = 0; r < 8; ++r) {
        int row = m0 + wave * 16 + rhi * 8 + r;
        if (row < M) {
            int c0 = n0 + (lane & 15);
            int c1 = c0 + 16;
            if (c0 < N) {
                float v = acc0[r];
                if (bias) v += bias[c0];
                if (act == 1)      v = fmaxf(v, 0.0f);
                else if (act == 2) v = (v > 20.0f) ? v : log1pf(__expf(v));
                C[(size_t)row * ldc + c0] = v;
            }
            if (c1 < N) {
                float v = acc1[r];
                if (bias) v += bias[c1];
                if (act == 1)      v = fmaxf(v, 0.0f);
                else if (act == 2) v = (v > 20.0f) ? v : log1pf(__expf(v));
                C[(size_t)row * ldc + c1] = v;
            }
        }
    }
}

// ---------------------------------------------------------------------------
// GCN helpers
// ---------------------------------------------------------------------------
__global__ void k_init_deg(float* deg) {
    int i = blockIdx.x * blockDim.x + threadIdx.x;
    if (i < Nn) deg[i] = 1.0f;                       // self-loop
}
__global__ void k_count_deg(const int* __restrict__ dst, float* deg) {
    int e = blockIdx.x * blockDim.x + threadIdx.x;
    if (e < Ec) atomAddF(&deg[dst[e]], 1.0f);
}
__global__ void k_dinv(float* deg) {
    int i = blockIdx.x * blockDim.x + threadIdx.x;
    if (i < Nn) deg[i] = rsqrtf(fmaxf(deg[i], 1.0f));
}
// agg[i,:] = XW[i,:] * dinv[i]^2   (self-loop term, also zero-initializes agg)
__global__ void k_agg_init(const float* __restrict__ XW,
                           const float* __restrict__ dinv,
                           float* __restrict__ agg) {
    int gid = blockIdx.x * blockDim.x + threadIdx.x;
    if (gid >= Nn * Hc) return;
    int i = gid >> 8;  // /Hc
    float dv = dinv[i];
    agg[gid] = XW[gid] * dv * dv;
}
// per-edge scatter: agg[dst] += XW[src] * dinv[src]*dinv[dst]  (4 cols/thread)
__global__ void k_scatter(const float* __restrict__ XW,
                          const int* __restrict__ src,
                          const int* __restrict__ dst,
                          const float* __restrict__ dinv,
                          float* __restrict__ agg) {
    int gid = blockIdx.x * blockDim.x + threadIdx.x;
    int e = gid >> 6;                 // 64 chunks of 4 cols each (Hc=256)
    if (e >= Ec) return;
    int c0 = (gid & 63) * 4;
    int s = src[e], d = dst[e];
    float coef = dinv[s] * dinv[d];
    const float4 xv = *reinterpret_cast<const float4*>(XW + (size_t)s * Hc + c0);
    float* op = agg + (size_t)d * Hc + c0;
    atomAddF(op + 0, xv.x * coef);
    atomAddF(op + 1, xv.y * coef);
    atomAddF(op + 2, xv.z * coef);
    atomAddF(op + 3, xv.w * coef);
}
__global__ void k_gcn_final(const float* __restrict__ agg,
                            const float* __restrict__ b,
                            float* __restrict__ h) {
    int gid = blockIdx.x * blockDim.x + threadIdx.x;
    if (gid >= Nn * Hc) return;
    int c = gid & (Hc - 1);
    h[gid] = fmaxf(agg[gid] + b[c], 0.0f);
}

// ---------------------------------------------------------------------------
// Mamba pieces
// ---------------------------------------------------------------------------
// causal depthwise conv1d (K=4) + silu. x lives in xz[:, 0:DIc] (stride 2*DIc)
__global__ void k_conv_silu(const float* __restrict__ xz,
                            const float* __restrict__ cw,
                            const float* __restrict__ cb,
                            float* __restrict__ xs) {
    int gid = blockIdx.x * blockDim.x + threadIdx.x;
    if (gid >= Nn * DIc) return;
    int t = gid / DIc, d = gid - t * DIc;
    float acc = cb[d];
    #pragma unroll
    for (int k = 0; k < 4; ++k) {
        int tt = t - 3 + k;
        if (tt >= 0) acc += xz[(size_t)tt * (2 * DIc) + d] * cw[d * 4 + k];
    }
    xs[gid] = acc / (1.0f + __expf(-acc));          // silu
}

__global__ void k_negexp(const float* __restrict__ alog, float* __restrict__ an) {
    int i = blockIdx.x * blockDim.x + threadIdx.x;
    if (i < DIc * DSc) an[i] = -__expf(alog[i]);
}

// Pass A: local scan per chunk with h0=0; also running product of a.
__global__ __launch_bounds__(512) void k_scan_a(
    const float* __restrict__ delta, const float* __restrict__ xs,
    const float* __restrict__ dtbc,  const float* __restrict__ an,
    float* __restrict__ ch, float* __restrict__ ca)
{
    int c = blockIdx.x, d = threadIdx.x;
    float A[DSc], hs[DSc], ap[DSc];
    #pragma unroll
    for (int s = 0; s < DSc; ++s) { A[s] = an[d * DSc + s]; hs[s] = 0.0f; ap[s] = 1.0f; }
    int t0 = c * CLEN;
    for (int t = t0; t < t0 + CLEN; ++t) {
        float dl = delta[(size_t)t * DIc + d];
        float xv = xs[(size_t)t * DIc + d];
        float dx = dl * xv;
        #pragma unroll
        for (int s = 0; s < DSc; ++s) {
            float a  = __expf(dl * A[s]);
            float bx = dx * dtbc[t * 48 + 16 + s];   // B columns
            hs[s] = a * hs[s] + bx;
            ap[s] *= a;
        }
    }
    size_t base = ((size_t)c * DIc + d) * DSc;
    #pragma unroll
    for (int s = 0; s < DSc; ++s) { ch[base + s] = hs[s]; ca[base + s] = ap[s]; }
}

// Pass B: serial combine over the 100 chunk summaries (tiny).
__global__ __launch_bounds__(512) void k_scan_b(
    const float* __restrict__ ch, const float* __restrict__ ca,
    float* __restrict__ ci)
{
    int d = threadIdx.x;
    float carry[DSc];
    #pragma unroll
    for (int s = 0; s < DSc; ++s) carry[s] = 0.0f;
    for (int c = 0; c < NCHUNK; ++c) {
        size_t base = ((size_t)c * DIc + d) * DSc;
        #pragma unroll
        for (int s = 0; s < DSc; ++s) {
            float old = carry[s];
            ci[base + s] = old;
            carry[s] = ca[base + s] * old + ch[base + s];
        }
    }
}

// Pass C: re-scan with correct carry, y = hs@C, fused (y + D*x)*silu(z).
// yout aliases the (dead) x-half of xz, stride 2*DIc.
__global__ __launch_bounds__(512) void k_scan_c(
    const float* __restrict__ delta, const float* __restrict__ xs,
    const float* __restrict__ dtbc,  const float* __restrict__ an,
    const float* __restrict__ ci,    const float* __restrict__ Dp,
    const float* __restrict__ xz,    float* __restrict__ yout)
{
    int c = blockIdx.x, d = threadIdx.x;
    float A[DSc], hs[DSc];
    size_t base = ((size_t)c * DIc + d) * DSc;
    #pragma unroll
    for (int s = 0; s < DSc; ++s) { A[s] = an[d * DSc + s]; hs[s] = ci[base + s]; }
    float dd = Dp[d];
    int t0 = c * CLEN;
    for (int t = t0; t < t0 + CLEN; ++t) {
        float dl = delta[(size_t)t * DIc + d];
        float xv = xs[(size_t)t * DIc + d];
        float dx = dl * xv;
        float y = 0.0f;
        #pragma unroll
        for (int s = 0; s < DSc; ++s) {
            float a = __expf(dl * A[s]);
            hs[s] = a * hs[s] + dx * dtbc[t * 48 + 16 + s];
            y += hs[s] * dtbc[t * 48 + 32 + s];      // C columns
        }
        float zv = xz[(size_t)t * (2 * DIc) + DIc + d];
        float sz = zv / (1.0f + __expf(-zv));        // silu(z)
        yout[(size_t)t * (2 * DIc) + d] = (y + dd * xv) * sz;
    }
}

// ---------------------------------------------------------------------------
// LayerNorm(h + hm) per row, accumulate mean-pool into gsum (wave32 per row)
// ---------------------------------------------------------------------------
__global__ void k_zero256(float* p) {
    int i = threadIdx.x;
    if (i < Hc) p[i] = 0.0f;
}
__global__ __launch_bounds__(256) void k_ln_pool(
    const float* __restrict__ h, const float* __restrict__ hm,
    const float* __restrict__ g, const float* __restrict__ b,
    float* __restrict__ gsum)
{
    int lane = threadIdx.x & 31;
    int row  = blockIdx.x * 8 + (threadIdx.x >> 5);
    if (row >= Nn) return;
    float v[8];
    float s = 0.0f;
    #pragma unroll
    for (int j = 0; j < 8; ++j) {
        int c = lane + 32 * j;
        v[j] = h[(size_t)row * Hc + c] + hm[(size_t)row * Hc + c];
        s += v[j];
    }
    #pragma unroll
    for (int m = 16; m >= 1; m >>= 1) s += __shfl_xor(s, m, 32);
    float mu = s * (1.0f / Hc);
    float vs = 0.0f;
    #pragma unroll
    for (int j = 0; j < 8; ++j) { float dv = v[j] - mu; vs += dv * dv; }
    #pragma unroll
    for (int m = 16; m >= 1; m >>= 1) vs += __shfl_xor(vs, m, 32);
    float rinv = rsqrtf(vs * (1.0f / Hc) + 1e-5f);
    #pragma unroll
    for (int j = 0; j < 8; ++j) {
        int c = lane + 32 * j;
        float nv = (v[j] - mu) * rinv * g[c] + b[c];
        atomAddF(&gsum[c], nv);
    }
}

// ---------------------------------------------------------------------------
// Head GEMV: out[j] = bias[j] + sum_h (gsum[h]/Nn) * W[h,j]
// ---------------------------------------------------------------------------
__global__ void k_head(const float* __restrict__ gsum,
                       const float* __restrict__ W,
                       const float* __restrict__ b,
                       float* __restrict__ out, int ncols)
{
    int j = blockIdx.x * blockDim.x + threadIdx.x;
    if (j >= ncols) return;
    float acc = b[j];
    const float invN = 1.0f / (float)Nn;
    #pragma unroll 8
    for (int hh = 0; hh < Hc; ++hh)
        acc += (gsum[hh] * invN) * W[(size_t)hh * ncols + j];
    out[j] = acc;
}

// ---------------------------------------------------------------------------
extern "C" void kernel_launch(void* const* d_in, const int* in_sizes, int n_in,
                              void* d_out, int out_size, void* d_ws, size_t ws_size,
                              hipStream_t stream)
{
    const float* nodef = (const float*)d_in[0];
    const int*   eidx  = (const int*)d_in[1];
    const float* g1w = (const float*)d_in[2];
    const float* g1b = (const float*)d_in[3];
    const float* g2w = (const float*)d_in[4];
    const float* g2b = (const float*)d_in[5];
    const float* inw = (const float*)d_in[6];
    const float* cw  = (const float*)d_in[7];
    const float* cb  = (const float*)d_in[8];
    const float* xpw = (const float*)d_in[9];
    const float* dtw = (const float*)d_in[10];
    const float* dtb = (const float*)d_in[11];
    const float* alg = (const float*)d_in[12];
    const float* Dp  = (const float*)d_in[13];
    const float* opw = (const float*)d_in[14];
    const float* lng = (const float*)d_in[15];
    const float* lnb = (const float*)d_in[16];
    const float* criW = (const float*)d_in[17]; const float* criB = (const float*)d_in[18];
    const float* higW = (const float*)d_in[19]; const float* higB = (const float*)d_in[20];
    const float* typW = (const float*)d_in[21]; const float* typB = (const float*)d_in[22];
    const float* p1W  = (const float*)d_in[23]; const float* p1B  = (const float*)d_in[24];
    const float* p2W  = (const float*)d_in[25]; const float* p2B  = (const float*)d_in[26];
    const float* depW = (const float*)d_in[27]; const float* depB = (const float*)d_in[28];
    const float* selW = (const float*)d_in[29]; const float* selB = (const float*)d_in[30];
    (void)in_sizes; (void)n_in; (void)out_size; (void)ws_size;

    float* out = (float*)d_out;
    float* ws  = (float*)d_ws;

    // ---- workspace carve-up (~119 MiB of f32) ----
    size_t off = 0;
    auto carve = [&](size_t n) { float* p = ws + off; off += (n + 255) & ~(size_t)255; return p; };
    float* dinv  = carve(Nn);
    float* B0    = carve((size_t)Nn * Hc);          // XW scratch
    float* B1    = carve((size_t)Nn * Hc);          // agg / hm
    float* B2    = carve((size_t)Nn * Hc);          // h1 / h2
    float* xz    = carve((size_t)Nn * 2 * DIc);     // in_proj out; x-half reused as y
    float* xs    = carve((size_t)Nn * DIc);         // conv+silu(x)
    float* dtbc  = carve((size_t)Nn * 48);          // [dt|B|C]
    float* delta = carve((size_t)Nn * DIc);
    float* aneg  = carve((size_t)DIc * DSc);
    float* ch    = carve((size_t)NCHUNK * DIc * DSc);
    float* ca    = carve((size_t)NCHUNK * DIc * DSc);
    float* ci    = carve((size_t)NCHUNK * DIc * DSc);
    float* gsum  = carve(Hc);

    const int* src = eidx;
    const int* dst = eidx + Ec;

    auto gemm = [&](const float* A, int lda, const float* B, int ldb,
                    const float* bias, float* C, int ldc,
                    int M, int Kd, int N, int act) {
        dim3 grid((M + GT_M - 1) / GT_M, (N + GT_N - 1) / GT_N);
        k_gemm<<<grid, dim3(128), 0, stream>>>(A, lda, B, ldb, bias, C, ldc, M, Kd, N, act);
    };

    // ---- degrees / symmetric normalization ----
    k_init_deg<<<(Nn + 255) / 256, 256, 0, stream>>>(dinv);
    k_count_deg<<<(Ec + 255) / 256, 256, 0, stream>>>(dst, dinv);
    k_dinv<<<(Nn + 255) / 256, 256, 0, stream>>>(dinv);

    // ---- GCN layer 1 ----
    gemm(nodef, Fc, g1w, Hc, nullptr, B0, Hc, Nn, Fc, Hc, 0);
    k_agg_init<<<(Nn * Hc + 255) / 256, 256, 0, stream>>>(B0, dinv, B1);
    k_scatter<<<((size_t)Ec * 64 + 255) / 256, 256, 0, stream>>>(B0, src, dst, dinv, B1);
    k_gcn_final<<<(Nn * Hc + 255) / 256, 256, 0, stream>>>(B1, g1b, B2);   // h1 -> B2

    // ---- GCN layer 2 ----
    gemm(B2, Hc, g2w, Hc, nullptr, B0, Hc, Nn, Hc, Hc, 0);
    k_agg_init<<<(Nn * Hc + 255) / 256, 256, 0, stream>>>(B0, dinv, B1);
    k_scatter<<<((size_t)Ec * 64 + 255) / 256, 256, 0, stream>>>(B0, src, dst, dinv, B1);
    k_gcn_final<<<(Nn * Hc + 255) / 256, 256, 0, stream>>>(B1, g2b, B2);   // h2 -> B2

    // ---- Mamba ----
    gemm(B2, Hc, inw, 2 * DIc, nullptr, xz, 2 * DIc, Nn, Hc, 2 * DIc, 0);  // in_proj
    k_conv_silu<<<(Nn * DIc + 255) / 256, 256, 0, stream>>>(xz, cw, cb, xs);
    gemm(xs, DIc, xpw, 48, nullptr, dtbc, 48, Nn, DIc, 48, 0);             // x_proj
    gemm(dtbc, 48, dtw, DIc, dtb, delta, DIc, Nn, 16, DIc, 2);             // dt_proj+softplus
    k_negexp<<<(DIc * DSc + 255) / 256, 256, 0, stream>>>(alg, aneg);
    k_scan_a<<<NCHUNK, 512, 0, stream>>>(delta, xs, dtbc, aneg, ch, ca);
    k_scan_b<<<1, 512, 0, stream>>>(ch, ca, ci);
    k_scan_c<<<NCHUNK, 512, 0, stream>>>(delta, xs, dtbc, aneg, ci, Dp, xz, xz);
    gemm(xz, 2 * DIc, opw, Hc, nullptr, B1, Hc, Nn, DIc, Hc, 0);           // out_proj -> hm

    // ---- LayerNorm(h + hm) + global mean pool ----
    k_zero256<<<1, 256, 0, stream>>>(gsum);
    k_ln_pool<<<(Nn + 7) / 8, 256, 0, stream>>>(B2, B1, lng, lnb, gsum);

    // ---- heads (concat: 1 + 4 + 3 + NS + NS + NS + NC = 30016) ----
    auto head = [&](const float* W, const float* b, float* o, int nc) {
        k_head<<<(nc + 63) / 64, 64, 0, stream>>>(gsum, W, b, o, nc);
    };
    head(criW, criB, out + 0, 1);
    head(higW, higB, out + 1, 4);
    head(typW, typB, out + 5, 3);
    head(p1W,  p1B,  out + 8, NSc);
    head(p2W,  p2B,  out + 8 + NSc, NSc);
    head(depW, depB, out + 8 + 2 * NSc, NSc);
    head(selW, selB, out + 8 + 3 * NSc, NCc);
}
